// MambaTransformerblock_20976620273904
// MI455X (gfx1250) — compile-verified, matrix operand order checked by reading, not run on hardware
//
#include <hip/hip_runtime.h>
#include <hip/hip_bf16.h>

// ---------------------------------------------------------------------------
// Mamba block forward for gfx1250 (CDNA5).
//  * Big GEMMs: bf16 WMMA (v_wmma_f32_16x16x32_bf16), f32 accumulate,
//    register-blocked 32x64 per wave (2x4 tiles) for operand reuse.
//  * Selective scan: exact chunked linear scan (partial / combine / emit),
//    16x more parallelism than a naive sequential scan.
// ---------------------------------------------------------------------------

typedef __bf16 bf16_t;
typedef __attribute__((ext_vector_type(16))) __bf16 v16bf;
typedef __attribute__((ext_vector_type(8)))  __bf16 v8bf;
typedef __attribute__((ext_vector_type(8)))  float  v8f;

#define NB      2
#define LSEQ    1024
#define DMODEL  1024
#define DI      2048             // dim_inner
#define NST     16               // d_state
#define RANK    64               // dt_rank
#define NROW    (NB * LSEQ)      // 2048 flattened (b,l) rows
#define XDBL_C  (RANK + 2 * NST) // 96
#define CHUNK   64               // scan chunk length
#define NCH     (LSEQ / CHUNK)   // 16 chunks per sequence

// ---------------------------------------------------------------------------
// elementwise cast fp32 -> bf16
__global__ void k_cast_bf16(const float* __restrict__ in, bf16_t* __restrict__ out, int n) {
    int i = blockIdx.x * blockDim.x + threadIdx.x;
    if (i < n) out[i] = (bf16_t)in[i];
}

// transpose-cast: in [rows, cols] fp32 -> out [cols, rows] bf16
__global__ void k_transpose_cast(const float* __restrict__ in, bf16_t* __restrict__ out,
                                 int rows, int cols) {
    int i = blockIdx.x * blockDim.x + threadIdx.x;
    if (i >= rows * cols) return;
    int c = i / rows;
    int r = i - c * rows;
    out[(size_t)c * rows + r] = (bf16_t)in[(size_t)r * cols + c];
}

// ---------------------------------------------------------------------------
// Register-blocked bf16 WMMA GEMM:
//   C[M,N] (f32) = A[M,K] (bf16 row-major) x B[K,N] (given as Bt[N,K] bf16)
// One wave computes a 32x64 strip = 2 (M) x 4 (N) tiles of 16x16.
// Each A fragment is reused 4x, each B fragment 2x per K step.
//
// Per ISA 7.12.2 (16-bit operand layout, B held column-per-lane):
//   lanes 0-15 : row/col = lane,    hold K = k0+{0..7} and k0+{16..23}
//   lanes 16-31: row/col = lane-16, hold K = k0+{8..15} and k0+{24..31}
// C/D f32 layout: VGPR r -> M=r (lanes 0-15), M=r+8 (lanes 16-31); N=lane&15.
__global__ void k_wmma_gemm(const bf16_t* __restrict__ A, const bf16_t* __restrict__ Bt,
                            float* __restrict__ C, int M, int N, int K) {
    const int lane    = threadIdx.x & 31;
    const int wave    = (blockIdx.x * blockDim.x + threadIdx.x) >> 5;
    const int stripsN = N >> 6;                    // 64-wide strips
    const int nStrips = (M >> 5) * stripsN;        // 32-tall strips
    if (wave >= nStrips) return;                   // wave-uniform; EXEC stays all-ones
    const int sm = wave / stripsN;
    const int sn = wave - sm * stripsN;
    const int r  = lane & 15;
    const int hi = lane >> 4;                      // K sub-group selector

    const bf16_t* Arow[2];
    const bf16_t* Brow[4];
#pragma unroll
    for (int i = 0; i < 2; ++i) Arow[i] = A  + (size_t)(sm * 32 + i * 16 + r) * K;
#pragma unroll
    for (int j = 0; j < 4; ++j) Brow[j] = Bt + (size_t)(sn * 64 + j * 16 + r) * K;

    v8f acc[2][4] = {};
    for (int k0 = 0; k0 < K; k0 += 32) {
        const int ka = k0 + hi * 8;
        __builtin_prefetch((const void*)(Arow[0] + ka + 512), 0, 3);
        __builtin_prefetch((const void*)(Brow[0] + ka + 512), 0, 3);

        v16bf av[2], bv[4];
#pragma unroll
        for (int i = 0; i < 2; ++i) {
            v8bf lo = *(const v8bf*)(Arow[i] + ka);
            v8bf hi8 = *(const v8bf*)(Arow[i] + ka + 16);
#pragma unroll
            for (int e = 0; e < 8; ++e) { av[i][e] = lo[e]; av[i][e + 8] = hi8[e]; }
        }
#pragma unroll
        for (int j = 0; j < 4; ++j) {
            v8bf lo = *(const v8bf*)(Brow[j] + ka);
            v8bf hi8 = *(const v8bf*)(Brow[j] + ka + 16);
#pragma unroll
            for (int e = 0; e < 8; ++e) { bv[j][e] = lo[e]; bv[j][e + 8] = hi8[e]; }
        }
#pragma unroll
        for (int i = 0; i < 2; ++i)
#pragma unroll
            for (int j = 0; j < 4; ++j)
                acc[i][j] = __builtin_amdgcn_wmma_f32_16x16x32_bf16(
                                false, av[i], false, bv[j], (short)0, acc[i][j],
                                false, false);
    }

#pragma unroll
    for (int i = 0; i < 2; ++i)
#pragma unroll
        for (int j = 0; j < 4; ++j) {
            float* Crow = C + (size_t)(sm * 32 + i * 16 + hi * 8) * N
                            + (sn * 64 + j * 16 + r);
#pragma unroll
            for (int e = 0; e < 8; ++e) Crow[(size_t)e * N] = acc[i][j][e];
        }
}

// ---------------------------------------------------------------------------
// causal depthwise conv1d (width 4) + bias + SiLU over the u-half of x_and_res
// xr: [NROW, 2*DI]; u_act: [NROW, DI]
__global__ void k_conv_silu(const float* __restrict__ xr, const float* __restrict__ cw,
                            const float* __restrict__ cb, float* __restrict__ u_act) {
    int i = blockIdx.x * blockDim.x + threadIdx.x;
    if (i >= NROW * DI) return;
    int row = i / DI;
    int d   = i - row * DI;
    int l   = row & (LSEQ - 1);
    float acc = cb[d];
#pragma unroll
    for (int j = 0; j < 4; ++j) {
        int ll = l - 3 + j;
        if (ll >= 0)
            acc += xr[(size_t)(row - 3 + j) * (2 * DI) + d] * cw[d * 4 + j];
    }
    u_act[i] = acc / (1.0f + __expf(-acc));   // SiLU
}

// ---------------------------------------------------------------------------
// x_dbl[NROW, 96] = u_act[NROW, DI] @ W_x[DI, 96]   (skinny; scalar dot)
__global__ void k_xdbl(const float* __restrict__ u, const float* __restrict__ Wx,
                       float* __restrict__ xd) {
    int i = blockIdx.x * blockDim.x + threadIdx.x;
    if (i >= NROW * XDBL_C) return;
    int row = i / XDBL_C;
    int c   = i - row * XDBL_C;
    const float* ur = u + (size_t)row * DI;
    float acc = 0.0f;
    for (int k = 0; k < DI; ++k) acc += ur[k] * Wx[(size_t)k * XDBL_C + c];
    xd[i] = acc;
}

// ---------------------------------------------------------------------------
// delta[NROW, DI] = softplus(x_dbl[:, :RANK] @ W_dt[RANK, DI] + b_dt)
__global__ void k_delta(const float* __restrict__ xd, const float* __restrict__ Wdt,
                        const float* __restrict__ bdt, float* __restrict__ delta) {
    int i = blockIdx.x * blockDim.x + threadIdx.x;
    if (i >= NROW * DI) return;
    int row = i / DI;
    int d   = i - row * DI;
    const float* dr = xd + (size_t)row * XDBL_C;
    float acc = bdt[d];
#pragma unroll 8
    for (int k = 0; k < RANK; ++k) acc += dr[k] * Wdt[(size_t)k * DI + d];
    delta[i] = (acc > 20.0f) ? acc : log1pf(__expf(acc));   // softplus
}

// ---------------------------------------------------------------------------
// Chunked selective scan (exact, linear-diagonal recurrence):
//   h_t = exp(dv_t*A_n) h_{t-1} + dv_t*u_t*B_t[n]
// Phase 1: per (b,d,chunk) compute zero-init partial state q[n] and sum(dv)
//          (prod of exp(dv*A) over the chunk == exp(A*sum_dv), exactly).
__global__ void k_scan_partial(const float* __restrict__ delta, const float* __restrict__ u,
                               const float* __restrict__ xd, const float* __restrict__ A_log,
                               float* __restrict__ qbuf, float* __restrict__ sdbuf) {
    int tid = blockIdx.x * blockDim.x + threadIdx.x;   // ((b*DI + d)*NCH + c)
    if (tid >= NB * DI * NCH) return;
    int c  = tid % NCH;
    int bd = tid / NCH;
    int d  = bd % DI;
    int b  = bd / DI;

    float An[NST], h[NST];
#pragma unroll
    for (int n = 0; n < NST; ++n) {
        An[n] = -__expf(A_log[(size_t)d * NST + n]);
        h[n]  = 0.0f;
    }
    float sd = 0.0f;
    for (int t = 0; t < CHUNK; ++t) {
        size_t row = (size_t)b * LSEQ + c * CHUNK + t;
        float dv = delta[row * DI + d];
        float uv = u[row * DI + d];
        const float* bc = xd + row * XDBL_C + RANK;
        float dbu = dv * uv;
        sd += dv;
#pragma unroll
        for (int n = 0; n < NST; ++n)
            h[n] = __expf(dv * An[n]) * h[n] + dbu * bc[n];
    }
    float* q = qbuf + (size_t)tid * NST;
#pragma unroll
    for (int n = 0; n < NST; ++n) q[n] = h[n];
    sdbuf[tid] = sd;
}

// Phase 2: sequential combine over the (few) chunks -> h_init per chunk.
__global__ void k_scan_combine(const float* __restrict__ A_log, const float* __restrict__ qbuf,
                               const float* __restrict__ sdbuf, float* __restrict__ hinit) {
    int tid = blockIdx.x * blockDim.x + threadIdx.x;   // (b*DI + d)
    if (tid >= NB * DI) return;
    int d = tid % DI;
    float An[NST], h[NST];
#pragma unroll
    for (int n = 0; n < NST; ++n) {
        An[n] = -__expf(A_log[(size_t)d * NST + n]);
        h[n]  = 0.0f;
    }
    for (int c = 0; c < NCH; ++c) {
        size_t base = ((size_t)tid * NCH + c) * NST;
        float sd = sdbuf[(size_t)tid * NCH + c];
#pragma unroll
        for (int n = 0; n < NST; ++n) {
            hinit[base + n] = h[n];
            h[n] = __expf(sd * An[n]) * h[n] + qbuf[base + n];
        }
    }
}

// Phase 3: per (b,d,chunk) re-run the recurrence from known h_init, emit
// gated output y directly as bf16 (+ D-skip + SiLU(res) gate).
__global__ void k_scan_emit(const float* __restrict__ delta, const float* __restrict__ u,
                            const float* __restrict__ xd, const float* __restrict__ xr,
                            const float* __restrict__ A_log, const float* __restrict__ Dvec,
                            const float* __restrict__ hinit, bf16_t* __restrict__ yb) {
    int tid = blockIdx.x * blockDim.x + threadIdx.x;   // ((b*DI + d)*NCH + c)
    if (tid >= NB * DI * NCH) return;
    int c  = tid % NCH;
    int bd = tid / NCH;
    int d  = bd % DI;
    int b  = bd / DI;

    float An[NST], h[NST];
#pragma unroll
    for (int n = 0; n < NST; ++n) {
        An[n] = -__expf(A_log[(size_t)d * NST + n]);
        h[n]  = hinit[(size_t)tid * NST + n];
    }
    const float Dd = Dvec[d];

    for (int t = 0; t < CHUNK; ++t) {
        size_t row = (size_t)b * LSEQ + c * CHUNK + t;
        float dv = delta[row * DI + d];
        float uv = u[row * DI + d];
        float rv = xr[row * (2 * DI) + DI + d];        // res (gate input)
        const float* bc = xd + row * XDBL_C + RANK;    // B[0..15], C[0..15]
        float dbu = dv * uv;
        float y = 0.0f;
#pragma unroll
        for (int n = 0; n < NST; ++n) {
            h[n] = __expf(dv * An[n]) * h[n] + dbu * bc[n];
            y += h[n] * bc[NST + n];
        }
        y += uv * Dd;
        float gate = rv / (1.0f + __expf(-rv));        // silu(res)
        yb[row * DI + d] = (bf16_t)(y * gate);
    }
}

// ---------------------------------------------------------------------------
extern "C" void kernel_launch(void* const* d_in, const int* in_sizes, int n_in,
                              void* d_out, int out_size, void* d_ws, size_t ws_size,
                              hipStream_t stream) {
    (void)in_sizes; (void)n_in; (void)out_size; (void)ws_size;

    const float* x     = (const float*)d_in[0];   // [2,1024,1024]
    const float* W_in  = (const float*)d_in[1];   // [1024, 4096]
    const float* cw    = (const float*)d_in[2];   // [2048, 4]
    const float* cb    = (const float*)d_in[3];   // [2048]
    const float* W_x   = (const float*)d_in[4];   // [2048, 96]
    const float* W_dt  = (const float*)d_in[5];   // [64, 2048]
    const float* b_dt  = (const float*)d_in[6];   // [2048]
    const float* A_log = (const float*)d_in[7];   // [2048, 16]
    const float* Dv    = (const float*)d_in[8];   // [2048]
    const float* W_out = (const float*)d_in[9];   // [2048, 1024]
    float* out = (float*)d_out;                   // [2,1024,1024]

    // workspace carve-out (256B-aligned slabs)
    char* ws = (char*)d_ws;
    size_t off = 0;
    auto carve = [&](size_t bytes) -> void* {
        void* p = ws + off;
        off = (off + bytes + 255) & ~(size_t)255;
        return p;
    };
    bf16_t* x_bf   = (bf16_t*)carve((size_t)NROW * DMODEL * sizeof(bf16_t));      // 4 MB
    bf16_t* WinT   = (bf16_t*)carve((size_t)(2 * DI) * DMODEL * sizeof(bf16_t));  // 8 MB
    bf16_t* WoutT  = (bf16_t*)carve((size_t)DMODEL * DI * sizeof(bf16_t));        // 4 MB
    float*  xr     = (float*) carve((size_t)NROW * 2 * DI * sizeof(float));       // 32 MB
    float*  u_act  = (float*) carve((size_t)NROW * DI * sizeof(float));           // 16 MB
    float*  xd     = (float*) carve((size_t)NROW * XDBL_C * sizeof(float));       // .75 MB
    float*  dlt    = (float*) carve((size_t)NROW * DI * sizeof(float));           // 16 MB
    bf16_t* y_bf   = (bf16_t*)carve((size_t)NROW * DI * sizeof(bf16_t));          // 8 MB
    float*  qbuf   = (float*) carve((size_t)NB * DI * NCH * NST * sizeof(float)); // 4 MB
    float*  sdbuf  = (float*) carve((size_t)NB * DI * NCH * sizeof(float));       // .25 MB
    float*  hinit  = (float*) carve((size_t)NB * DI * NCH * NST * sizeof(float)); // 4 MB

    const int T = 256;
    auto blocks = [](long n, int t) { return (unsigned)((n + t - 1) / t); };

    // 1) casts / transposes of GEMM operands into bf16
    k_cast_bf16<<<blocks((long)NROW * DMODEL, T), T, 0, stream>>>(x, x_bf, NROW * DMODEL);
    k_transpose_cast<<<blocks((long)DMODEL * 2 * DI, T), T, 0, stream>>>(W_in,  WinT,  DMODEL, 2 * DI);
    k_transpose_cast<<<blocks((long)DI * DMODEL, T), T, 0, stream>>>(W_out, WoutT, DI, DMODEL);

    // 2) in_proj: xr[2048,4096] = x_bf[2048,1024] @ W_in   (WMMA, 32x64/wave)
    {
        long waves = (long)(NROW / 32) * ((2 * DI) / 64);     // 4096 waves
        k_wmma_gemm<<<blocks(waves * 32, T), T, 0, stream>>>(
            x_bf, WinT, xr, NROW, 2 * DI, DMODEL);
    }

    // 3) causal depthwise conv + bias + SiLU
    k_conv_silu<<<blocks((long)NROW * DI, T), T, 0, stream>>>(xr, cw, cb, u_act);

    // 4) x_dbl = u_act @ W_x
    k_xdbl<<<blocks((long)NROW * XDBL_C, T), T, 0, stream>>>(u_act, W_x, xd);

    // 5) delta = softplus(delta_r @ W_dt + b_dt)
    k_delta<<<blocks((long)NROW * DI, T), T, 0, stream>>>(xd, W_dt, b_dt, dlt);

    // 6) chunked selective scan (partial -> combine -> emit) -> y_bf
    k_scan_partial<<<blocks((long)NB * DI * NCH, T), T, 0, stream>>>(
        dlt, u_act, xd, A_log, qbuf, sdbuf);
    k_scan_combine<<<blocks((long)NB * DI, T), T, 0, stream>>>(
        A_log, qbuf, sdbuf, hinit);
    k_scan_emit<<<blocks((long)NB * DI * NCH, T), T, 0, stream>>>(
        dlt, u_act, xd, xr, A_log, Dv, hinit, y_bf);

    // 7) out_proj: out[2048,1024] = y_bf[2048,2048] @ W_out   (WMMA, 32x64/wave)
    {
        long waves = (long)(NROW / 32) * (DMODEL / 64);       // 1024 waves
        k_wmma_gemm<<<blocks(waves * 32, T), T, 0, stream>>>(
            y_bf, WoutT, out, NROW, DMODEL, DI);
    }
}
